// SS2D_6081673691144
// MI455X (gfx1250) — compile-verified, matrix operand order checked by reading, not run on hardware
//
#include <hip/hip_runtime.h>
#include <hip/hip_bf16.h>
#include <math.h>

// Problem constants (from reference)
#define BB 2
#define CC 128
#define HH 64
#define WW 64
#define LL 4096      // H*W
#define EE 128
#define NN 16
#define KK 4
#define RR 8
#define DBL 40       // R + 2N
#define CH 64        // scan chunk length
#define NCH 64       // number of chunks (LL / CH)

typedef __attribute__((ext_vector_type(16))) _Float16 v16h;
typedef __attribute__((ext_vector_type(8)))  float    v8f;

// ---------------------------------------------------------------------------
// WMMA helpers (CDNA5 16x16x32 f16 -> f32, layouts per ISA 7.12.2)
// ---------------------------------------------------------------------------
__device__ __forceinline__ v8f wmma32(v16h a, v16h b, v8f c) {
  return __builtin_amdgcn_wmma_f32_16x16x32_f16(
      /*neg_a=*/false, a, /*neg_b=*/false, b,
      /*c_mod=*/(short)0, c, /*reuse_a=*/false, /*reuse_b=*/false);
}

// A fragment from LDS f16 tile: rows m = lane%16,
// per-lane element t -> K = (t/8)*16 + 8*(lane/16) + t%8
__device__ __forceinline__ v16h frag_a_h(const _Float16* src, int ld, int kk, int lane) {
  int m = lane & 15;
  int hf = (lane >> 4) << 3;
  v16h a;
#pragma unroll
  for (int t = 0; t < 16; ++t) {
    int k = ((t >> 3) << 4) + hf + (t & 7);
    a[t] = src[m * ld + kk + k];
  }
  return a;
}

// B fragment from pre-packed f16 weights: fragment (kt,nt) stored as
// 32 lanes x 16 contiguous halves -> one aligned 32B load per lane.
__device__ __forceinline__ v16h frag_b_p(const _Float16* Wp, int ntiles,
                                         int kt, int nt, int lane) {
  return *(const v16h*)(Wp + (((size_t)kt * ntiles + nt) * 32 + lane) * 16);
}

// seq index <-> spatial index for direction dir (involution; col scans are a
// 6-bit halves swap because H == W == 64)
__device__ __forceinline__ int seq2sp(int dir, int s) {
  int t = (dir & 1) ? (LL - 1 - s) : s;
  if (dir >= 2) t = ((t & 63) << 6) | (t >> 6);
  return t;
}

__device__ __forceinline__ float silu(float v) {
  return v / (1.f + __expf(-v));
}

// ---------------------------------------------------------------------------
// K0: repack an f32 row-major weight (K x Ncols, leading dim Nld) into f16
// WMMA-B fragment order, padding columns to Npad (multiple of 16).
// ---------------------------------------------------------------------------
__global__ void __launch_bounds__(256)
k_pack(const float* __restrict__ W, _Float16* __restrict__ dst,
       int K, int Nld, int Ncols, int Npad) {
  int idx = blockIdx.x * 256 + threadIdx.x;
  int total = (K >> 5) * (Npad >> 4) * 512;
  if (idx >= total) return;
  int t    = idx & 15;
  int lane = (idx >> 4) & 31;
  int tile = idx >> 9;
  int ntiles = Npad >> 4;
  int nt = tile % ntiles, kt = tile / ntiles;
  int k = kt * 32 + ((t >> 3) << 4) + ((lane >> 4) << 3) + (t & 7);
  int n = nt * 16 + (lane & 15);
  dst[idx] = (n < Ncols) ? (_Float16)W[(size_t)k * Nld + n] : (_Float16)0.f;
}

// ---------------------------------------------------------------------------
// K1: LayerNorm over C + inproj GEMM (128 -> 256) -> xp, z
// grid (L/16, B), block 256 (8 waves, 2 N-tiles each)
// ---------------------------------------------------------------------------
__global__ void __launch_bounds__(256)
k_ln_inproj(const float* __restrict__ x, const float* __restrict__ g,
            const float* __restrict__ bta, const _Float16* __restrict__ Wp,
            float* __restrict__ xp, float* __restrict__ z) {
  __shared__ float    sx[16][CC];
  __shared__ _Float16 sa[16][CC];
  int b = blockIdx.y, tile = blockIdx.x, tid = threadIdx.x;

  for (int i = tid; i < 16 * CC; i += 256) {
    int t = i >> 7, c = i & 127;
    sx[t][c] = x[((size_t)b * CC + c) * LL + tile * 16 + t];  // xs[b,l,c]=x[b,c,l]
  }
  __syncthreads();

  if (tid < 16) {
    float mu = 0.f;
    for (int c = 0; c < CC; ++c) mu += sx[tid][c];
    mu *= (1.f / CC);
    float var = 0.f;
    for (int c = 0; c < CC; ++c) { float d = sx[tid][c] - mu; var += d * d; }
    float inv = rsqrtf(var * (1.f / CC) + 1e-5f);
    for (int c = 0; c < CC; ++c)
      sa[tid][c] = (_Float16)((sx[tid][c] - mu) * inv * g[c] + bta[c]);
  }
  __syncthreads();

  int wave = tid >> 5, lane = tid & 31;
#pragma unroll
  for (int j = 0; j < 2; ++j) {
    int nt = wave * 2 + j;
    v8f acc = {};
#pragma unroll
    for (int kt = 0; kt < 4; ++kt)
      acc = wmma32(frag_a_h(&sa[0][0], CC, kt * 32, lane),
                   frag_b_p(Wp, 16, kt, nt, lane), acc);
    int n = nt * 16 + (lane & 15), mh = (lane >> 4) << 3;
    float* dst = (n < EE) ? xp : z;
    int nc = n & (EE - 1);
#pragma unroll
    for (int r = 0; r < 8; ++r) {
      int tok = tile * 16 + mh + r;
      dst[((size_t)b * LL + tok) * EE + nc] = acc[r];
    }
  }
}

// ---------------------------------------------------------------------------
// K2: gather direction sequence + Win GEMM (128 -> 256) -> xin, zi
// ---------------------------------------------------------------------------
__global__ void __launch_bounds__(256)
k_win(const float* __restrict__ xp, const _Float16* __restrict__ Wp,
      float* __restrict__ xin, float* __restrict__ zi, int dir) {
  __shared__ _Float16 sa[16][EE];
  int b = blockIdx.y, tile = blockIdx.x, tid = threadIdx.x;

  for (int i = tid; i < 16 * EE; i += 256) {
    int t = i >> 7, c = i & 127;
    int sp = seq2sp(dir, tile * 16 + t);
    sa[t][c] = (_Float16)xp[((size_t)b * LL + sp) * EE + c];
  }
  __syncthreads();

  int wave = tid >> 5, lane = tid & 31;
#pragma unroll
  for (int j = 0; j < 2; ++j) {
    int nt = wave * 2 + j;
    v8f acc = {};
#pragma unroll
    for (int kt = 0; kt < 4; ++kt)
      acc = wmma32(frag_a_h(&sa[0][0], EE, kt * 32, lane),
                   frag_b_p(Wp, 16, kt, nt, lane), acc);
    int n = nt * 16 + (lane & 15), mh = (lane >> 4) << 3;
    float* dst = (n < EE) ? xin : zi;
    int nc = n & (EE - 1);
#pragma unroll
    for (int r = 0; r < 8; ++r) {
      int tok = tile * 16 + mh + r;
      dst[((size_t)b * LL + tok) * EE + nc] = acc[r];
    }
  }
}

// ---------------------------------------------------------------------------
// K3a: depthwise causal conv (K=4) + bias + SiLU -> xc
// ---------------------------------------------------------------------------
__global__ void __launch_bounds__(256)
k_conv(const float* __restrict__ xin, const float* __restrict__ cw,
       const float* __restrict__ cb, float* __restrict__ xc) {
  int idx = blockIdx.x * 256 + threadIdx.x;
  if (idx >= BB * LL * EE) return;
  int e = idx & (EE - 1);
  int s = (idx >> 7) & (LL - 1);
  int b = idx >> 19;
  float acc = cb[e];
#pragma unroll
  for (int k = 0; k < KK; ++k) {
    int ss = s - (KK - 1) + k;
    if (ss >= 0) acc += cw[e * KK + k] * xin[((size_t)b * LL + ss) * EE + e];
  }
  xc[idx] = silu(acc);
}

// ---------------------------------------------------------------------------
// K3b: xdbl = xc @ Wx (128 -> 40, padded to 48)  [dt(8) | B(16) | C(16)]
// grid (L/16, B), block 96 (3 waves = 3 N-tiles, last half-masked)
// ---------------------------------------------------------------------------
__global__ void __launch_bounds__(96)
k_xdbl(const float* __restrict__ xc, const _Float16* __restrict__ Wp,
       float* __restrict__ xdbl) {
  __shared__ _Float16 sa[16][EE];
  int b = blockIdx.y, tile = blockIdx.x, tid = threadIdx.x;

  for (int i = tid; i < 16 * EE; i += 96) {
    int t = i >> 7, c = i & 127;
    sa[t][c] = (_Float16)xc[((size_t)b * LL + tile * 16 + t) * EE + c];
  }
  __syncthreads();

  int wave = tid >> 5, lane = tid & 31;
  v8f acc = {};
#pragma unroll
  for (int kt = 0; kt < 4; ++kt)
    acc = wmma32(frag_a_h(&sa[0][0], EE, kt * 32, lane),
                 frag_b_p(Wp, 3, kt, wave, lane), acc);
  int n = wave * 16 + (lane & 15), mh = (lane >> 4) << 3;
  if (n < DBL) {
#pragma unroll
    for (int r = 0; r < 8; ++r) {
      int tok = tile * 16 + mh + r;
      xdbl[((size_t)b * LL + tok) * DBL + n] = acc[r];
    }
  }
}

// ---------------------------------------------------------------------------
// K3c: dtp = softplus(dt8 @ Wdt + bdt)   (8 -> 128, tiny: pure VALU)
// ---------------------------------------------------------------------------
__global__ void __launch_bounds__(256)
k_dt(const float* __restrict__ xdbl, const float* __restrict__ Wdt,
     const float* __restrict__ bdt, float* __restrict__ dtp) {
  int idx = blockIdx.x * 256 + threadIdx.x;
  if (idx >= BB * LL * EE) return;
  int e = idx & (EE - 1);
  size_t tok = (size_t)(idx >> 7);
  float a = bdt[e];
#pragma unroll
  for (int r = 0; r < RR; ++r)
    a += xdbl[tok * DBL + r] * Wdt[r * EE + e];
  dtp[idx] = (a > 20.f) ? a : log1pf(__expf(a));
}

// ---------------------------------------------------------------------------
// K4a: chunked scan pass A — per chunk: P = prod(a), h_loc with zero carry.
// grid (NCH, E/16, B), block 256 (8 waves x 2 channels x 16 states)
// ---------------------------------------------------------------------------
__global__ void __launch_bounds__(256)
k_scanA(const float* __restrict__ dtp, const float* __restrict__ xc,
        const float* __restrict__ xdbl, const float* __restrict__ Alog,
        float* __restrict__ Pc, float* __restrict__ Hc) {
  int chunk = blockIdx.x, b = blockIdx.z;
  int wave = threadIdx.x >> 5, lane = threadIdx.x & 31;
  int n = lane & 15, cl = lane >> 4;
  int e = blockIdx.y * 16 + wave * 2 + cl;

  float A = -__expf(Alog[e * NN + n]);
  float P = 1.f, h = 0.f;
  int s0 = chunk * CH;
  for (int j = 0; j < CH; ++j) {
    size_t tok = (size_t)b * LL + s0 + j;
    float dt  = dtp[tok * EE + e];
    float xcv = xc[tok * EE + e];
    float Bm  = xdbl[tok * DBL + RR + n];
    float a   = __expf(dt * A);
    h = a * h + dt * Bm * xcv;
    P *= a;
  }
  size_t o = (((size_t)b * NCH + chunk) * EE + e) * NN + n;
  Pc[o] = P;
  Hc[o] = h;
}

// ---------------------------------------------------------------------------
// K4b: chunked scan pass B — serial prefix over 64 chunks per (b,e,n).
// ---------------------------------------------------------------------------
__global__ void __launch_bounds__(256)
k_scanB(const float* __restrict__ Pc, const float* __restrict__ Hc,
        float* __restrict__ Hs) {
  int idx = blockIdx.x * 256 + threadIdx.x;
  if (idx >= BB * EE * NN) return;
  int en = idx % (EE * NN);
  int b  = idx / (EE * NN);
  float carry = 0.f;
  for (int c = 0; c < NCH; ++c) {
    size_t o = ((size_t)b * NCH + c) * (EE * NN) + en;
    Hs[o] = carry;
    carry = Pc[o] * carry + Hc[o];
  }
}

// ---------------------------------------------------------------------------
// K4c: chunked scan pass C — replay chunk from carry-in, contract with C_t
// (16-lane shfl_xor tree), fuse +D*xc and *silu(z).
// ---------------------------------------------------------------------------
__global__ void __launch_bounds__(256)
k_scanC(const float* __restrict__ dtp, const float* __restrict__ xc,
        const float* __restrict__ xdbl, const float* __restrict__ Alog,
        const float* __restrict__ Dp, const float* __restrict__ zi,
        const float* __restrict__ Hs, float* __restrict__ yact) {
  int chunk = blockIdx.x, b = blockIdx.z;
  int wave = threadIdx.x >> 5, lane = threadIdx.x & 31;
  int n = lane & 15, cl = lane >> 4;
  int e = blockIdx.y * 16 + wave * 2 + cl;

  float A  = -__expf(Alog[e * NN + n]);
  float Dv = Dp[e];
  float h  = Hs[(((size_t)b * NCH + chunk) * EE + e) * NN + n];
  int s0 = chunk * CH;
  for (int j = 0; j < CH; ++j) {
    size_t tok = (size_t)b * LL + s0 + j;
    float dt  = dtp[tok * EE + e];
    float xcv = xc[tok * EE + e];
    float Bm  = xdbl[tok * DBL + RR + n];
    float Cm  = xdbl[tok * DBL + RR + NN + n];
    float a   = __expf(dt * A);
    h = a * h + dt * Bm * xcv;
    float p = h * Cm;
    p += __shfl_xor(p, 1, 32);
    p += __shfl_xor(p, 2, 32);
    p += __shfl_xor(p, 4, 32);
    p += __shfl_xor(p, 8, 32);
    if (n == 0) {
      float zv = zi[tok * EE + e];
      yact[tok * EE + e] = (p + Dv * xcv) * silu(zv);
    }
  }
}

// ---------------------------------------------------------------------------
// K5: yo = yact @ Wout (128x128) WMMA, scatter(-add) to spatial order
// grid (L/16, B), block 256 (8 waves = 8 N-tiles)
// ---------------------------------------------------------------------------
__global__ void __launch_bounds__(256)
k_wout(const float* __restrict__ yact, const _Float16* __restrict__ Wp,
       float* __restrict__ ycomb, int dir) {
  __shared__ _Float16 sa[16][EE];
  int b = blockIdx.y, tile = blockIdx.x, tid = threadIdx.x;

  for (int i = tid; i < 16 * EE; i += 256) {
    int t = i >> 7, c = i & 127;
    sa[t][c] = (_Float16)yact[((size_t)b * LL + tile * 16 + t) * EE + c];
  }
  __syncthreads();

  int wave = tid >> 5, lane = tid & 31;
  v8f acc = {};
#pragma unroll
  for (int kt = 0; kt < 4; ++kt)
    acc = wmma32(frag_a_h(&sa[0][0], EE, kt * 32, lane),
                 frag_b_p(Wp, 8, kt, wave, lane), acc);
  int n = wave * 16 + (lane & 15), mh = (lane >> 4) << 3;
#pragma unroll
  for (int r = 0; r < 8; ++r) {
    int s = tile * 16 + mh + r;
    int sp = seq2sp(dir, s);
    size_t o = ((size_t)b * LL + sp) * EE + n;
    if (dir == 0) ycomb[o] = acc[r];
    else          ycomb[o] += acc[r];
  }
}

// ---------------------------------------------------------------------------
// K6: y = ycomb/4 * silu(z); outproj GEMM (128x128) WMMA; LayerNorm +
// residual; transpose store to (B,C,H,W).
// ---------------------------------------------------------------------------
__global__ void __launch_bounds__(256)
k_final(const float* __restrict__ ycomb, const float* __restrict__ z,
        const _Float16* __restrict__ Wp, const float* __restrict__ og,
        const float* __restrict__ ob, const float* __restrict__ x,
        float* __restrict__ out) {
  __shared__ _Float16 sa[16][EE];
  __shared__ float    sy[16][CC];
  int b = blockIdx.y, tile = blockIdx.x, tid = threadIdx.x;

  for (int i = tid; i < 16 * EE; i += 256) {
    int t = i >> 7, c = i & 127;
    size_t tok = (size_t)b * LL + tile * 16 + t;
    float zv = z[tok * EE + c];
    sa[t][c] = (_Float16)(ycomb[tok * EE + c] * 0.25f * silu(zv));
  }
  __syncthreads();

  int wave = tid >> 5, lane = tid & 31;
  v8f acc = {};
#pragma unroll
  for (int kt = 0; kt < 4; ++kt)
    acc = wmma32(frag_a_h(&sa[0][0], EE, kt * 32, lane),
                 frag_b_p(Wp, 8, kt, wave, lane), acc);
  int n = wave * 16 + (lane & 15), mh = (lane >> 4) << 3;
#pragma unroll
  for (int r = 0; r < 8; ++r) sy[mh + r][n] = acc[r];
  __syncthreads();

  if (tid < 16) {
    float mu = 0.f;
    for (int c = 0; c < CC; ++c) mu += sy[tid][c];
    mu *= (1.f / CC);
    float var = 0.f;
    for (int c = 0; c < CC; ++c) { float d = sy[tid][c] - mu; var += d * d; }
    float inv = rsqrtf(var * (1.f / CC) + 1e-5f);
    int l = tile * 16 + tid;
    for (int c = 0; c < CC; ++c) {
      float o = (sy[tid][c] - mu) * inv * og[c] + ob[c];
      o += x[((size_t)b * CC + c) * LL + l];     // residual = xs
      out[((size_t)b * CC + c) * LL + l] = o;    // (B,C,H,W)
    }
  }
}

// ---------------------------------------------------------------------------
extern "C" void kernel_launch(void* const* d_in, const int* in_sizes, int n_in,
                              void* d_out, int out_size, void* d_ws, size_t ws_size,
                              hipStream_t stream) {
  (void)in_sizes; (void)n_in; (void)out_size; (void)ws_size;
  const float* x      = (const float*)d_in[0];
  const float* ng     = (const float*)d_in[1];
  const float* nb     = (const float*)d_in[2];
  const float* Wip    = (const float*)d_in[3];
  const float* Wop    = (const float*)d_in[4];
  const float* og     = (const float*)d_in[5];
  const float* ob     = (const float*)d_in[6];
  const float* mWin   = (const float*)d_in[7];
  const float* mconvw = (const float*)d_in[8];
  const float* mconvb = (const float*)d_in[9];
  const float* mWx    = (const float*)d_in[10];
  const float* mWdt   = (const float*)d_in[11];
  const float* mbdt   = (const float*)d_in[12];
  const float* mAlog  = (const float*)d_in[13];
  const float* mD     = (const float*)d_in[14];
  const float* mWout  = (const float*)d_in[15];

  // ---- float workspace ----
  float* ws = (float*)d_ws;
  const size_t TE  = (size_t)BB * LL * EE;       // 1,048,576
  const size_t SC  = (size_t)BB * NCH * EE * NN; // 262,144 (scan chunk state)
  float* xp    = ws;
  float* z     = xp    + TE;
  float* xin   = z     + TE;
  float* zi    = xin   + TE;
  float* xc    = zi    + TE;
  float* dtp   = xc    + TE;
  float* yact  = dtp   + TE;
  float* ycomb = yact  + TE;
  float* xdbl  = ycomb + TE;                      // BB*LL*DBL = 327,680
  float* Pc    = xdbl  + (size_t)BB * LL * DBL;   // SC
  float* Hc    = Pc    + SC;                      // SC
  float* Hs    = Hc    + SC;                      // SC
  float* fend  = Hs    + SC;

  // ---- packed f16 weights (WMMA B-fragment order), ~540 KB ----
  _Float16* hp = (_Float16*)fend;
  const size_t P_WIN  = (size_t)128 * 256;  // 32768 halves per 128x256 matrix
  const size_t P_WX   = (size_t)128 * 48;   // 6144 halves (40 padded to 48)
  const size_t P_W128 = (size_t)128 * 128;  // 16384 halves per 128x128 matrix
  _Float16* pWip  = hp;
  _Float16* pWop  = pWip + P_WIN;
  _Float16* pWin  = pWop + P_W128;          // x4 directions
  _Float16* pWx   = pWin + 4 * P_WIN;       // x4
  _Float16* pWout = pWx  + 4 * P_WX;        // x4

  // ---- pack all weights (cheap, fully parallel) ----
  k_pack<<<(int)((P_WIN  + 255) / 256), 256, 0, stream>>>(Wip, pWip, 128, 256, 256, 256);
  k_pack<<<(int)((P_W128 + 255) / 256), 256, 0, stream>>>(Wop, pWop, 128, 128, 128, 128);
  for (int d = 0; d < 4; ++d) {
    k_pack<<<(int)((P_WIN  + 255) / 256), 256, 0, stream>>>(
        mWin  + (size_t)d * 128 * 256, pWin  + (size_t)d * P_WIN,  128, 256, 256, 256);
    k_pack<<<(int)((P_WX   + 255) / 256), 256, 0, stream>>>(
        mWx   + (size_t)d * 128 * DBL, pWx   + (size_t)d * P_WX,   128, DBL, DBL, 48);
    k_pack<<<(int)((P_W128 + 255) / 256), 256, 0, stream>>>(
        mWout + (size_t)d * 128 * 128, pWout + (size_t)d * P_W128, 128, 128, 128, 128);
  }

  dim3 gTile(LL / 16, BB);
  dim3 gScan(NCH, EE / 16, BB);
  int nElem = BB * LL * EE;

  k_ln_inproj<<<gTile, 256, 0, stream>>>(x, ng, nb, pWip, xp, z);

  for (int d = 0; d < 4; ++d) {
    k_win  <<<gTile, 256, 0, stream>>>(xp, pWin + (size_t)d * P_WIN, xin, zi, d);
    k_conv <<<(nElem + 255) / 256, 256, 0, stream>>>(xin, mconvw + d * EE * KK,
                                                     mconvb + d * EE, xc);
    k_xdbl <<<gTile, 96, 0, stream>>>(xc, pWx + (size_t)d * P_WX, xdbl);
    k_dt   <<<(nElem + 255) / 256, 256, 0, stream>>>(xdbl, mWdt + d * RR * EE,
                                                     mbdt + d * EE, dtp);
    k_scanA<<<gScan, 256, 0, stream>>>(dtp, xc, xdbl, mAlog + d * EE * NN, Pc, Hc);
    k_scanB<<<(BB * EE * NN + 255) / 256, 256, 0, stream>>>(Pc, Hc, Hs);
    k_scanC<<<gScan, 256, 0, stream>>>(dtp, xc, xdbl, mAlog + d * EE * NN,
                                       mD + d * EE, zi, Hs, yact);
    k_wout <<<gTile, 256, 0, stream>>>(yact, pWout + (size_t)d * P_W128, ycomb, d);
  }

  k_final<<<gTile, 256, 0, stream>>>(ycomb, z, pWop, og, ob, x, (float*)d_out);
}
